// Transformer_1108101562695
// MI455X (gfx1250) — matrix operand order repaired and checked by reading, not compiled, not run on hardware
//
#include <hip/hip_runtime.h>
#include <stdint.h>

// ---------------------------------------------------------------------------
// CDNA5 (gfx1250) transformer forward, round 4 (final polish).
//  - bf16 WMMA (f32 acc); weights pre-transposed at convert time.
//  - GEMM: 64x256 block, 8 waves, wave tile 32x64 (8 WMMA/K-step),
//    double-buffered LDS, register prefetch, 1 barrier per K-step.
//  - Flash attention (S^T formulation), pre-transposed V buffer:
//    no LDS, no barriers; inner loop = global b128 loads + WMMA + softmax.
//  - Native __bf16 <-> float casts (hardware cvt) instead of bit-twiddling.
// ---------------------------------------------------------------------------

typedef __bf16 bf16_t;
typedef __attribute__((ext_vector_type(16))) __bf16 v16bf;
typedef __attribute__((ext_vector_type(8)))  __bf16 v8bf;
typedef __attribute__((ext_vector_type(8)))  float  v8f;

union V16 { v16bf v; v8bf h[2]; };

__device__ __forceinline__ bf16_t f2bf(float x) { return (bf16_t)x; }
__device__ __forceinline__ float  bf2f(bf16_t x) { return (float)x; }

__device__ __forceinline__ v8bf ld8(const bf16_t* p) { return *(const v8bf*)p; }

__device__ __forceinline__ v8f wmma_bf16(v16bf a, v16bf b, v8f c) {
  return __builtin_amdgcn_wmma_f32_16x16x32_bf16(false, a, false, b, (short)0, c,
                                                 false, false);
}

// ---------------------------------------------------------------------------
// weight convert + transpose: in[K][N] f32 -> out[N][K] bf16 (32x32 LDS tiles)
// ---------------------------------------------------------------------------
__global__ __launch_bounds__(256) void cvt_transpose(const float* __restrict__ in,
                                                     bf16_t* __restrict__ out,
                                                     int K, int N) {
  __shared__ float tile[32][33];
  const int n0 = blockIdx.x * 32, k0 = blockIdx.y * 32;
  const int tx = threadIdx.x & 31, ty = threadIdx.x >> 5;   // 32 x 8
#pragma unroll
  for (int i = 0; i < 4; ++i)
    tile[ty + i * 8][tx] = in[(size_t)(k0 + ty + i * 8) * N + n0 + tx];
  __syncthreads();
#pragma unroll
  for (int i = 0; i < 4; ++i)
    out[(size_t)(n0 + ty + i * 8) * K + k0 + tx] = f2bf(tile[tx][ty + i * 8]);
}

// ---------------------------------------------------------------------------
// V transpose: qkv[8192][2304] (V block) -> vt[b*12+h][64][1024] bf16
// ---------------------------------------------------------------------------
__global__ __launch_bounds__(256) void vt_transpose(const bf16_t* __restrict__ qkv,
                                                    bf16_t* __restrict__ vt) {
  __shared__ bf16_t tile[32][33];
  const int n0 = blockIdx.x * 32;          // seq position tile
  const int d0 = blockIdx.y * 32;          // head-dim tile
  const int bh = blockIdx.z;               // b*12 + h
  const int bb = bh / 12, h = bh % 12;
  const int tx = threadIdx.x & 31, ty = threadIdx.x >> 5;
#pragma unroll
  for (int i = 0; i < 4; ++i)
    tile[ty + i * 8][tx] =
        qkv[(size_t)(bb * 1024 + n0 + ty + i * 8) * 2304 + 1536 + h * 64 + d0 + tx];
  __syncthreads();
#pragma unroll
  for (int i = 0; i < 4; ++i)
    vt[((size_t)bh * 64 + d0 + ty + i * 8) * 1024 + n0 + tx] = tile[tx][ty + i * 8];
}

// ---------------------------------------------------------------------------
// LayerNorm: one wave per row, 8 rows per 256-thread block
// ---------------------------------------------------------------------------
template <bool BF16OUT>
__global__ __launch_bounds__(256) void ln_kernel(const float* __restrict__ x,
                                                 const float* __restrict__ w,
                                                 const float* __restrict__ b,
                                                 void* out, int D) {
  const int wave = threadIdx.x >> 5, lane = threadIdx.x & 31;
  const int row = blockIdx.x * 8 + wave;
  const float* xr = x + (size_t)row * D;
  float ls = 0.f, lq = 0.f;
  for (int i = lane; i < D; i += 32) { float v = xr[i]; ls += v; lq += v * v; }
#pragma unroll
  for (int m = 16; m; m >>= 1) {
    ls += __shfl_xor(ls, m, 32);
    lq += __shfl_xor(lq, m, 32);
  }
  const float mu = ls / D;
  const float rstd = rsqrtf(lq / D - mu * mu + 1e-5f);
  for (int i = lane; i < D; i += 32) {
    float v = (xr[i] - mu) * rstd * w[i] + b[i];
    if (BF16OUT) ((bf16_t*)out)[(size_t)row * D + i] = f2bf(v);
    else         ((float*)out)[(size_t)row * D + i]  = v;
  }
}

// ---------------------------------------------------------------------------
// GEMM: C[M,N] = A[M,K](bf16) @ W[K,N] with W supplied TRANSPOSED Wt[N][K].
// ---------------------------------------------------------------------------
#define G_LDA 40
#define G_LDB 40

template <bool BIAS, bool RES, bool GELU_, bool BF16OUT>
__global__ __launch_bounds__(256) void gemm_bf16(const bf16_t* __restrict__ A,
                                                 const bf16_t* __restrict__ Wt,
                                                 const float* __restrict__ bias,
                                                 const float* res, float* outf,
                                                 bf16_t* outb, int M, int N, int K) {
  __shared__ bf16_t As[2][64 * G_LDA];
  __shared__ bf16_t Bs[2][256 * G_LDB];
  const int tid = threadIdx.x;
  const int lane = tid & 31, wave = tid >> 5;
  const int wm = wave >> 2, wn = wave & 3;
  const int lm = lane & 15, lh = lane >> 4;
  const int mbase = blockIdx.y * 64;
  const int nbase = blockIdx.x * 256;

  const int arow = tid >> 2;          // 0..63
  const int acol = (tid & 3) * 8;     // 0..24
  v8f c[2][4] = {};
  v8bf ra, rb[4];

  const bf16_t* Ap = A + (size_t)(mbase + arow) * K + acol;
#define LOAD_TILES(KT)                                                        \
  {                                                                           \
    ra = ld8(Ap + (KT));                                                      \
    _Pragma("unroll") for (int i = 0; i < 4; ++i) {                           \
      int cix = tid + i * 256;                                                \
      rb[i] = ld8(Wt + (size_t)(nbase + (cix >> 2)) * K + (KT) + (cix & 3) * 8); \
    }                                                                         \
  }
#define STORE_TILES(BUF)                                                      \
  {                                                                           \
    *(v8bf*)&As[BUF][arow * G_LDA + acol] = ra;                               \
    _Pragma("unroll") for (int i = 0; i < 4; ++i) {                           \
      int cix = tid + i * 256;                                                \
      *(v8bf*)&Bs[BUF][(cix >> 2) * G_LDB + (cix & 3) * 8] = rb[i];           \
    }                                                                         \
  }

  LOAD_TILES(0);
  STORE_TILES(0);
  __syncthreads();

  int cur = 0;
  for (int kt = 0; kt < K; kt += 32) {
    const bool more = (kt + 32) < K;
    if (more) LOAD_TILES(kt + 32);

    V16 af[2], bfr[4];
    const int kb = lh * 8;
#pragma unroll
    for (int mt = 0; mt < 2; ++mt) {
      const int r0 = wm * 32 + mt * 16 + lm;
      af[mt].h[0] = ld8(&As[cur][r0 * G_LDA + kb]);
      af[mt].h[1] = ld8(&As[cur][r0 * G_LDA + kb + 16]);
    }
#pragma unroll
    for (int nt = 0; nt < 4; ++nt) {
      const int n0 = wn * 64 + nt * 16 + lm;
      bfr[nt].h[0] = ld8(&Bs[cur][n0 * G_LDB + lh * 8]);
      bfr[nt].h[1] = ld8(&Bs[cur][n0 * G_LDB + lh * 8 + 16]);
    }
#pragma unroll
    for (int mt = 0; mt < 2; ++mt)
#pragma unroll
      for (int nt = 0; nt < 4; ++nt)
        c[mt][nt] = wmma_bf16(af[mt].v, bfr[nt].v, c[mt][nt]);

    if (more) {
      STORE_TILES(cur ^ 1);
      __syncthreads();
      cur ^= 1;
    }
  }

#pragma unroll
  for (int mt = 0; mt < 2; ++mt)
#pragma unroll
    for (int nt = 0; nt < 4; ++nt) {
      const int col = nbase + wn * 64 + nt * 16 + lm;
      const float bv = BIAS ? bias[col] : 0.f;
#pragma unroll
      for (int r = 0; r < 8; ++r) {
        const int row = mbase + wm * 32 + mt * 16 + lh * 8 + r;
        float v = c[mt][nt][r] + bv;
        if (GELU_) v = 0.5f * v * (1.f + erff(v * 0.70710678118654752f));
        if (RES) v += res[(size_t)row * N + col];
        if (BF16OUT) outb[(size_t)row * N + col] = f2bf(v);
        else         outf[(size_t)row * N + col] = v;
      }
    }
#undef LOAD_TILES
#undef STORE_TILES
}

// ---------------------------------------------------------------------------
// Flash attention, S^T formulation, LDS-free / barrier-free.
// ---------------------------------------------------------------------------
__global__ __launch_bounds__(256) void flash_attn(const bf16_t* __restrict__ qkv,
                                                  const bf16_t* __restrict__ vt,
                                                  bf16_t* __restrict__ out) {
  const int tid = threadIdx.x, lane = tid & 31, wave = tid >> 5;
  const int lm = lane & 15, lh = lane >> 4;
  const int h = blockIdx.y, bb = blockIdx.z;
  const int LDQ = 2304;
  const size_t rowQ0 = (size_t)bb * 1024 + blockIdx.x * 128 + wave * 16;
  const int colQ = h * 64, colK = 768 + h * 64;
  const bf16_t* vtp = vt + ((size_t)(bb * 12 + h)) * 64 * 1024;

  // Q as B-fragments of Q^T, pre-scaled by DH^-0.5 = 0.125 (exact in bf16).
  V16 bq0, bq1;
  {
    const bf16_t* qp = qkv + (rowQ0 + lm) * LDQ + colQ;
    bq0.h[0] = ld8(qp + lh * 8);      bq0.h[1] = ld8(qp + lh * 8 + 16);
    bq1.h[0] = ld8(qp + 32 + lh * 8); bq1.h[1] = ld8(qp + 32 + lh * 8 + 16);
#pragma unroll
    for (int j = 0; j < 16; ++j) {
      bq0.v[j] = f2bf(0.125f * bf2f(bq0.v[j]));
      bq1.v[j] = f2bf(0.125f * bf2f(bq1.v[j]));
    }
  }

  v8f o[4] = {};
  float mrun = -1e30f, lrun = 0.f;

  for (int kb0 = 0; kb0 < 1024; kb0 += 32) {
    // S^T tiles: A = K rows (16 keys x 32 d), B = Q^T. Two key tiles.
    v8f s0 = {}, s1 = {};
    {
      V16 ak;
      const bf16_t* kp0 = qkv + ((size_t)bb * 1024 + kb0 + lm) * LDQ + colK;
      ak.h[0] = ld8(kp0 + lh * 8);      ak.h[1] = ld8(kp0 + lh * 8 + 16);
      s0 = wmma_bf16(ak.v, bq0.v, s0);
      ak.h[0] = ld8(kp0 + 32 + lh * 8); ak.h[1] = ld8(kp0 + 32 + lh * 8 + 16);
      s0 = wmma_bf16(ak.v, bq1.v, s0);
      const bf16_t* kp1 = kp0 + 16 * LDQ;
      ak.h[0] = ld8(kp1 + lh * 8);      ak.h[1] = ld8(kp1 + lh * 8 + 16);
      s1 = wmma_bf16(ak.v, bq0.v, s1);
      ak.h[0] = ld8(kp1 + 32 + lh * 8); ak.h[1] = ld8(kp1 + 32 + lh * 8 + 16);
      s1 = wmma_bf16(ak.v, bq1.v, s1);
    }
    // online softmax for query q = lane&15: 16 in-lane scores + 1 shuffle
    float mloc = fmaxf(s0[0], s0[1]);
#pragma unroll
    for (int j = 2; j < 8; ++j) mloc = fmaxf(mloc, s0[j]);
#pragma unroll
    for (int j = 0; j < 8; ++j) mloc = fmaxf(mloc, s1[j]);
    mloc = fmaxf(mloc, __shfl_xor(mloc, 16, 32));
    const float mnew = fmaxf(mrun, mloc);
    const float alpha = __expf(mrun - mnew);
    V16 pa;
    float ps = 0.f;
#pragma unroll
    for (int j = 0; j < 8; ++j) {
      float p0 = __expf(s0[j] - mnew);
      float p1 = __expf(s1[j] - mnew);
      ps += p0 + p1;
      pa.v[j]     = f2bf(p0);   // keys 8*lh + j        (A-frag elems 0..7)
      pa.v[8 + j] = f2bf(p1);   // keys 16 + 8*lh + j   (A-frag elems 8..15)
    }
    ps += __shfl_xor(ps, 16, 32);
    lrun = lrun * alpha + ps;
    mrun = mnew;
    // rescale O (rows of O are q = 8*lh + r; alpha lives in lane q)
#pragma unroll
    for (int r = 0; r < 8; ++r) {
      const float ar = __shfl(alpha, lh * 8 + r, 32);
#pragma unroll
      for (int dt = 0; dt < 4; ++dt) o[dt][r] *= ar;
    }
    // O += P @ V  (V from pre-transposed global buffer, contiguous per lane)
#pragma unroll
    for (int dt = 0; dt < 4; ++dt) {
      V16 vb;
      const bf16_t* vp = vtp + (size_t)(dt * 16 + lm) * 1024 + kb0 + lh * 8;
      vb.h[0] = ld8(vp);
      vb.h[1] = ld8(vp + 16);
      o[dt] = wmma_bf16(pa.v, vb.v, o[dt]);
    }
  }
  // normalize and store bf16
#pragma unroll
  for (int r = 0; r < 8; ++r) {
    const float inv = 1.0f / __shfl(lrun, lh * 8 + r, 32);
    const size_t row = rowQ0 + lh * 8 + r;
#pragma unroll
    for (int dt = 0; dt < 4; ++dt)
      out[row * 768 + h * 64 + dt * 16 + lm] = f2bf(o[dt][r] * inv);
  }
}

// ---------------------------------------------------------------------------
// Host driver
// ---------------------------------------------------------------------------
extern "C" void kernel_launch(void* const* d_in, const int* in_sizes, int n_in,
                              void* d_out, int out_size, void* d_ws, size_t ws_size,
                              hipStream_t stream) {
  (void)in_sizes; (void)n_in; (void)out_size; (void)ws_size;
  const float* x    = (const float*)d_in[0];
  const float* ln1w = (const float*)d_in[1];
  const float* ln1b = (const float*)d_in[2];
  const float* wqkv = (const float*)d_in[3];
  const float* wo   = (const float*)d_in[4];
  const float* bo   = (const float*)d_in[5];
  const float* ln2w = (const float*)d_in[6];
  const float* ln2b = (const float*)d_in[7];
  const float* w1   = (const float*)d_in[8];
  const float* b1   = (const float*)d_in[9];
  const float* w2   = (const float*)d_in[10];
  const float* b2   = (const float*)d_in[11];
  const float* lnfw = (const float*)d_in[12];
  const float* lnfb = (const float*)d_in[13];

  const int Mr = 8 * 1024;
  const int D = 768, DF = 3072, TRI = 2304;

  char* ws = (char*)d_ws;
  size_t off = 0;
  auto alloc = [&](size_t bytes) -> void* {
    void* p = ws + off;
    off += (bytes + 255) & ~(size_t)255;
    return p;
  };
  float*  x_ws    = (float*) alloc((size_t)Mr * D * 4);
  bf16_t* h_bf    = (bf16_t*)alloc((size_t)Mr * D * 2);
  bf16_t* qkv_bf  = (bf16_t*)alloc((size_t)Mr * TRI * 2);
  bf16_t* attn_bf = (bf16_t*)alloc((size_t)Mr * D * 2);
  bf16_t* m1_bf   = (bf16_t*)alloc((size_t)Mr * DF * 2);
  bf16_t* vt_bf   = (bf16_t*)alloc((size_t)96 * 64 * 1024 * 2);
  bf16_t* wqkv_bt = (bf16_t*)alloc((size_t)D * TRI * 2);
  bf16_t* wo_bt   = (bf16_t*)alloc((size_t)D * D * 2);
  bf16_t* w1_bt   = (bf16_t*)alloc((size_t)D * DF * 2);
  bf16_t* w2_bt   = (bf16_t*)alloc((size_t)DF * D * 2);

  const dim3 blk256(256);
  for (int L = 0; L < 12; ++L) {
    const float* xin = (L == 0) ? x : x_ws;
    cvt_transpose<<<dim3(TRI/32, D/32), blk256, 0, stream>>>(wqkv + (size_t)L*D*TRI, wqkv_bt, D,  TRI);
    cvt_transpose<<<dim3(D/32,   D/32), blk256, 0, stream>>>(wo   + (size_t)L*D*D,   wo_bt,   D,  D);
    cvt_transpose<<<dim3(DF/32,  D/32), blk256, 0, stream>>>(w1   + (size_t)L*D*DF,  w1_bt,   D,  DF);
    cvt_transpose<<<dim3(D/32,  DF/32), blk256, 0, stream>>>(w2   + (size_t)L*DF*D,  w2_bt,   DF, D);

    ln_kernel<true><<<dim3(Mr/8), blk256, 0, stream>>>(xin, ln1w + L*D, ln1b + L*D, h_bf, D);
    gemm_bf16<false,false,false,true><<<dim3(TRI/256, Mr/64), blk256, 0, stream>>>(
        h_bf, wqkv_bt, nullptr, nullptr, nullptr, qkv_bf, Mr, TRI, D);
    vt_transpose<<<dim3(32, 2, 96), blk256, 0, stream>>>(qkv_bf, vt_bf);
    flash_attn<<<dim3(8, 12, 8), blk256, 0, stream>>>(qkv_bf, vt_bf, attn_bf);
    gemm_bf16<true,true,false,false><<<dim3(D/256, Mr/64), blk256, 0, stream>>>(
        attn_bf, wo_bt, bo + L*D, xin, x_ws, nullptr, Mr, D, D);
    ln_kernel<true><<<dim3(Mr/8), blk256, 0, stream>>>(x_ws, ln2w + L*D, ln2b + L*D, h_bf, D);
    gemm_bf16<true,false,true,true><<<dim3(DF/256, Mr/64), blk256, 0, stream>>>(
        h_bf, w1_bt, b1 + (size_t)L*DF, nullptr, nullptr, m1_bf, Mr, DF, D);
    gemm_bf16<true,true,false,false><<<dim3(D/256, Mr/64), blk256, 0, stream>>>(
        m1_bf, w2_bt, b2 + L*D, x_ws, x_ws, nullptr, Mr, D, DF);
  }
  ln_kernel<false><<<dim3(Mr/8), blk256, 0, stream>>>(x_ws, lnfw, lnfb, (float*)d_out, D);
}
